// CGB_37288906064501
// MI455X (gfx1250) — matrix operand order repaired and checked by reading, not compiled, not run on hardware
//
#include <hip/hip_runtime.h>

// ---------------- problem constants ----------------
#define NPTS 8192
#define KNN  16
#define CIN  64
#define CC   256
#define CS   32            // CC / share_planes
#define NK   (NPTS*KNN)    // 131072
#define EPS  1e-5f

// ---------------- WMMA / TDM vector types ----------------
typedef __attribute__((ext_vector_type(2)))  float        v2f;
typedef __attribute__((ext_vector_type(8)))  float        v8f;
typedef __attribute__((ext_vector_type(16))) __bf16       v16bf;
typedef __attribute__((ext_vector_type(4)))  unsigned int v4u;
typedef __attribute__((ext_vector_type(8)))  int          v8i_t;
typedef __attribute__((ext_vector_type(4)))  int          v4i_t;

// =====================================================================
// Tensor Data Mover: async 2-D tile load Global -> LDS (D# groups 0/1,
// groups 2/3 zero => 2-D tensor). Tracked by TENSORcnt.
//   data_size_code: 0=1B, 1=2B, 2=4B
//   tile_w/tile_h: tile dims (elements);  tensor_w/h: tensor dims
//   stride_elems : row stride of tensor (elements)
// =====================================================================
__device__ __forceinline__ void tdm_load_2d_to_lds(void* lds_dst, const void* gsrc,
                                                   unsigned data_size_code,
                                                   unsigned tile_w, unsigned tile_h,
                                                   unsigned tensor_w, unsigned tensor_h,
                                                   unsigned stride_elems) {
    unsigned lds_off = (unsigned)(unsigned long long)lds_dst;   // addr[31:0] == LDS offset
    unsigned long long ga = (unsigned long long)gsrc;
    v4u g0;
    g0.x = 1u;                                                  // count=1, user descriptor
    g0.y = lds_off;                                             // D#.lds_addr
    g0.z = (unsigned)ga;                                        // global_addr[31:0]
    g0.w = (unsigned)((ga >> 32) & 0x1FFFFFFull) | 0x80000000u; // addr[56:32] | type=2
    v8i_t g1;
    g1[0] = (int)(data_size_code << 16);                        // wg_mask=0, data_size
    g1[1] = (int)((tensor_w & 0xFFFFu) << 16);                  // tensor_dim0[15:0]
    g1[2] = (int)((tensor_w >> 16) | ((tensor_h & 0xFFFFu) << 16));
    g1[3] = (int)((tensor_h >> 16) | ((tile_w & 0xFFFFu) << 16));
    g1[4] = (int)(tile_h & 0xFFFFu);                            // tile_dim1 (tile_dim2=0)
    g1[5] = (int)stride_elems;                                  // tensor_dim0_stride[31:0]
    g1[6] = 0;
    g1[7] = 0;
    v4i_t z4 = {0, 0, 0, 0};
    v8i_t z8 = {0, 0, 0, 0, 0, 0, 0, 0};
    __builtin_amdgcn_tensor_load_to_lds(g0, g1, z4, z4, z8, 0);
}

// =====================================================================
// GEMM (fp32, WMMA 16x16x4): C = A[M,K]@B[K,N] (+bias)
// 4 waves / block share one TDM-staged 16-column B tile in LDS.
// grid = (M/64, N/16), block = 128
// =====================================================================
__global__ __launch_bounds__(128)
void gemm_f32_wmma(const float* __restrict__ A, const float* __restrict__ B,
                   const float* __restrict__ bias, float* __restrict__ Cmat,
                   int M, int Nn, int Kk) {
    __shared__ float sB[256 * 16];            // up to K=256 rows x 16 cols
    int lane = threadIdx.x & 31;
    int wv = threadIdx.x >> 5;
    int m0 = (blockIdx.x * 4 + wv) * 16, n0 = blockIdx.y * 16;
    if (wv == 0) {
        tdm_load_2d_to_lds(sB, B + n0, 2u, 16u, (unsigned)Kk,
                           (unsigned)Nn, (unsigned)Kk, (unsigned)Nn);
        __builtin_amdgcn_s_wait_tensorcnt(0);
    }
    __syncthreads();

    int row = m0 + (lane & 15);
    int khalf = (lane >> 4) * 2;              // lanes 0-15: K {0,1}; 16-31: K {2,3}
    int cl = lane & 15;
    const float* Arow = A + (size_t)row * Kk;
    v8f acc = {};
    for (int k0 = 0; k0 < Kk; k0 += 4) {
        __builtin_prefetch(Arow + k0 + 64, 0, 0);
        v2f a, b;
        a.x = Arow[k0 + khalf];
        a.y = Arow[k0 + khalf + 1];
        b.x = sB[(k0 + khalf) * 16 + cl];
        b.y = sB[(k0 + khalf + 1) * 16 + cl];
        acc = __builtin_amdgcn_wmma_f32_16x16x4_f32(
            false, a, false, b, (short)0, acc, false, false);
    }
    int col = n0 + cl;
    int mb = m0 + (lane >> 4) * 8;
    #pragma unroll
    for (int r = 0; r < 8; ++r) {
        float v = acc[r];
        if (bias) v += bias[col];
        Cmat[(size_t)(mb + r) * Nn + col] = v;
    }
}

// =====================================================================
// GEMM (bf16 A/B, f32 acc, WMMA 16x16x32), TDM-staged B tile in LDS.
// grid = (M/64, N/16), block = 128
// =====================================================================
__global__ __launch_bounds__(128)
void gemm_bf16_wmma(const __bf16* __restrict__ A, const __bf16* __restrict__ B,
                    const float* __restrict__ bias, float* __restrict__ Cmat,
                    int M, int Nn, int Kk) {
    __shared__ __bf16 sB[256 * 16];           // up to K=256 rows x 16 cols
    int lane = threadIdx.x & 31;
    int wv = threadIdx.x >> 5;
    int m0 = (blockIdx.x * 4 + wv) * 16, n0 = blockIdx.y * 16;
    if (wv == 0) {
        tdm_load_2d_to_lds(sB, B + n0, 1u, 16u, (unsigned)Kk,
                           (unsigned)Nn, (unsigned)Kk, (unsigned)Nn);
        __builtin_amdgcn_s_wait_tensorcnt(0);
    }
    __syncthreads();

    int row = m0 + (lane & 15);
    int kh = (lane >> 4) * 16;                // A: lanes<16 hold K 0-15, else 16-31
    int cl = lane & 15;
    int rb = (lane >> 4) * 16;                // B: lanes<16 rows 0-15, else rows 16-31
    const __bf16* Arow = A + (size_t)row * Kk;
    v8f acc = {};
    for (int k0 = 0; k0 < Kk; k0 += 32) {
        __builtin_prefetch(Arow + k0 + 128, 0, 0);
        v16bf a, b;
        #pragma unroll
        for (int i = 0; i < 8; ++i) a[i] = Arow[k0 + kh + i];
        #pragma unroll
        for (int i = 0; i < 8; ++i) a[8 + i] = Arow[k0 + kh + 8 + i];
        #pragma unroll
        for (int i = 0; i < 16; ++i) b[i] = sB[(k0 + rb + i) * 16 + cl];
        acc = __builtin_amdgcn_wmma_f32_16x16x32_bf16(
            false, a, false, b, (short)0, acc, false, false);
    }
    int col = n0 + cl;
    int mb = m0 + (lane >> 4) * 8;
    #pragma unroll
    for (int r = 0; r < 8; ++r) {
        float v = acc[r];
        if (bias) v += bias[col];
        Cmat[(size_t)(mb + r) * Nn + col] = v;
    }
}

// =====================================================================
// column mean / inv-std over rows (training BN stats), 1 block per column
// =====================================================================
__global__ __launch_bounds__(256)
void colstats(const float* __restrict__ X, float* __restrict__ mu,
              float* __restrict__ rstd, int rows, int cols) {
    int c = blockIdx.x;
    float s = 0.f, ss = 0.f;
    for (int r = threadIdx.x; r < rows; r += 256) {
        float v = X[(size_t)r * cols + c];
        s += v; ss += v * v;
    }
    __shared__ float sh[256], sh2[256];
    sh[threadIdx.x] = s; sh2[threadIdx.x] = ss;
    __syncthreads();
    for (int off = 128; off > 0; off >>= 1) {
        if (threadIdx.x < off) {
            sh[threadIdx.x]  += sh[threadIdx.x + off];
            sh2[threadIdx.x] += sh2[threadIdx.x + off];
        }
        __syncthreads();
    }
    if (threadIdx.x == 0) {
        float m = sh[0] / rows;
        mu[c] = m;
        rstd[c] = rsqrtf(fmaxf(sh2[0] / rows - m * m, 0.f) + EPS);
    }
}

__global__ __launch_bounds__(256)
void finalize_stats(const float* __restrict__ sum, const float* __restrict__ sumsq,
                    float* __restrict__ mu, float* __restrict__ rstd, int rows) {
    int c = threadIdx.x;
    float m = sum[c] / rows;
    mu[c] = m;
    rstd[c] = rsqrtf(fmaxf(sumsq[c] / rows - m * m, 0.f) + EPS);
}

// =====================================================================
// elementwise BN apply (+optional relu)
// =====================================================================
__global__ __launch_bounds__(256)
void bn_apply(const float* __restrict__ X, float* __restrict__ Y,
              const float* __restrict__ mu, const float* __restrict__ rstd,
              const float* __restrict__ g, const float* __restrict__ b,
              int total, int cols, int do_relu) {
    int i = blockIdx.x * 256 + threadIdx.x;
    if (i >= total) return;
    int c = i % cols;
    float v = (X[i] - mu[c]) * rstd[c] * g[c] + b[c];
    Y[i] = do_relu ? fmaxf(v, 0.f) : v;
}

// =====================================================================
// brute-force KNN top-16: 1 wave per query, LDS-staged candidate chunks
// =====================================================================
#define KCHUNK 2048
__global__ __launch_bounds__(256)
void knn_kernel(const float* __restrict__ p, int* __restrict__ idx) {
    __shared__ float sx[KCHUNK], sy[KCHUNK], sz[KCHUNK];
    int wave = threadIdx.x >> 5, lane = threadIdx.x & 31;
    int q = blockIdx.x * 8 + wave;
    float qx = p[q * 3 + 0], qy = p[q * 3 + 1], qz = p[q * 3 + 2];
    float dbest[KNN]; int ibest[KNN];
    #pragma unroll
    for (int i = 0; i < KNN; ++i) { dbest[i] = 3.4e38f; ibest[i] = 0x7fffffff; }
    for (int base = 0; base < NPTS; base += KCHUNK) {
        __syncthreads();
        for (int t = threadIdx.x; t < KCHUNK; t += 256) {
            sx[t] = p[(base + t) * 3 + 0];
            sy[t] = p[(base + t) * 3 + 1];
            sz[t] = p[(base + t) * 3 + 2];
        }
        __syncthreads();
        for (int c = lane; c < KCHUNK; c += 32) {
            float dx = sx[c] - qx, dy = sy[c] - qy, dz = sz[c] - qz;
            float d = dx * dx + dy * dy + dz * dz;
            if (d < dbest[KNN - 1]) {
                int pos = KNN - 1;
                while (pos > 0 && dbest[pos - 1] > d) {
                    dbest[pos] = dbest[pos - 1]; ibest[pos] = ibest[pos - 1]; --pos;
                }
                dbest[pos] = d; ibest[pos] = base + c;
            }
        }
    }
    // 16-round wave-min merge of 32 sorted per-lane lists
    int head = 0;
    for (int r = 0; r < KNN; ++r) {
        float d = (head < KNN) ? dbest[head] : 3.4e38f;
        int   id = (head < KNN) ? ibest[head] : 0x7fffffff;
        int   sl = lane;
        #pragma unroll
        for (int off = 16; off > 0; off >>= 1) {
            float od = __shfl_xor(d, off, 32);
            int   oi = __shfl_xor(id, off, 32);
            int   ol = __shfl_xor(sl, off, 32);
            if (od < d || (od == d && oi < id)) { d = od; id = oi; sl = ol; }
        }
        if (lane == 0) idx[q * KNN + r] = id;
        if (lane == sl) ++head;
    }
}

// =====================================================================
// pr1 = (p[idx]-p) @ Wp1 + bp1   -> [NK,3]
// =====================================================================
__global__ __launch_bounds__(256)
void pr1_kernel(const float* __restrict__ p, const int* __restrict__ idx,
                const float* __restrict__ Wp1, const float* __restrict__ bp1,
                float* __restrict__ pr) {
    int t = blockIdx.x * 256 + threadIdx.x;
    if (t >= NK) return;
    int n = t >> 4, j = idx[t];
    float rx = p[j * 3 + 0] - p[n * 3 + 0];
    float ry = p[j * 3 + 1] - p[n * 3 + 1];
    float rz = p[j * 3 + 2] - p[n * 3 + 2];
    #pragma unroll
    for (int c = 0; c < 3; ++c)
        pr[t * 3 + c] = rx * Wp1[c] + ry * Wp1[3 + c] + rz * Wp1[6 + c] + bp1[c];
}

// =====================================================================
// on-the-fly per-channel sums of w0 = xk[idx] - xq + (prA @ Wp2 + bp2)
// =====================================================================
__global__ __launch_bounds__(256)
void w0_stats(const float* __restrict__ xk, const float* __restrict__ xq,
              const int* __restrict__ idx, const float* __restrict__ prA,
              const float* __restrict__ Wp2, const float* __restrict__ bp2,
              float* __restrict__ sum, float* __restrict__ sumsq, int ppb) {
    int c = threadIdx.x;
    int p0 = blockIdx.x * ppb;
    float w0 = Wp2[c], w1 = Wp2[CC + c], w2 = Wp2[2 * CC + c], bb = bp2[c];
    float s = 0.f, ss = 0.f;
    for (int t = 0; t < ppb; ++t) {
        int pr = p0 + t, n = pr >> 4, j = idx[pr];
        float a0 = prA[pr * 3], a1 = prA[pr * 3 + 1], a2 = prA[pr * 3 + 2];
        float v = xk[(size_t)j * CC + c] - xq[(size_t)n * CC + c]
                + a0 * w0 + a1 * w1 + a2 * w2 + bb;
        s += v; ss += v * v;
    }
    atomicAdd(&sum[c], s);
    atomicAdd(&sumsq[c], ss);
}

// =====================================================================
// materialize relu(bn(w0)) as bf16 [NK,256]
// =====================================================================
__global__ __launch_bounds__(256)
void wbn_kernel(const float* __restrict__ xk, const float* __restrict__ xq,
                const int* __restrict__ idx, const float* __restrict__ prA,
                const float* __restrict__ Wp2, const float* __restrict__ bp2,
                const float* __restrict__ mu, const float* __restrict__ rstd,
                const float* __restrict__ g, const float* __restrict__ b,
                __bf16* __restrict__ out) {
    int pr = blockIdx.x, c = threadIdx.x;
    int n = pr >> 4, j = idx[pr];
    float a0 = prA[pr * 3], a1 = prA[pr * 3 + 1], a2 = prA[pr * 3 + 2];
    float v = xk[(size_t)j * CC + c] - xq[(size_t)n * CC + c]
            + a0 * Wp2[c] + a1 * Wp2[CC + c] + a2 * Wp2[2 * CC + c] + bp2[c];
    v = fmaxf((v - mu[c]) * rstd[c] * g[c] + b[c], 0.f);
    out[(size_t)pr * CC + c] = (__bf16)v;
}

__global__ __launch_bounds__(256)
void cvt_bf16(const float* __restrict__ in, __bf16* __restrict__ out, int n) {
    int i = blockIdx.x * 256 + threadIdx.x;
    if (i < n) out[i] = (__bf16)in[i];
}

// =====================================================================
// softmax over neighbors K for w [N,K,32], one wave per point (lane = cs)
// =====================================================================
__global__ __launch_bounds__(32)
void softmax_k(float* __restrict__ w) {
    int n = blockIdx.x, cs = threadIdx.x;
    float m = -3.4e38f;
    float e[KNN];
    #pragma unroll
    for (int k = 0; k < KNN; ++k) m = fmaxf(m, w[(size_t)(n * KNN + k) * CS + cs]);
    float s = 0.f;
    #pragma unroll
    for (int k = 0; k < KNN; ++k) {
        e[k] = expf(w[(size_t)(n * KNN + k) * CS + cs] - m);
        s += e[k];
    }
    float inv = 1.f / s;
    #pragma unroll
    for (int k = 0; k < KNN; ++k) w[(size_t)(n * KNN + k) * CS + cs] = e[k] * inv;
}

// =====================================================================
// y[n,c] = sum_k (xv[idx[n,k],c] + p_r(n,k,c)) * w[n,k,c%32]
// =====================================================================
__global__ __launch_bounds__(256)
void y_kernel(const float* __restrict__ xv, const int* __restrict__ idx,
              const float* __restrict__ prA, const float* __restrict__ Wp2,
              const float* __restrict__ bp2, const float* __restrict__ w,
              float* __restrict__ y) {
    int n = blockIdx.x, c = threadIdx.x;
    float w0 = Wp2[c], w1 = Wp2[CC + c], w2 = Wp2[2 * CC + c], bb = bp2[c];
    float acc = 0.f;
    #pragma unroll
    for (int k = 0; k < KNN; ++k) {
        int t = n * KNN + k, j = idx[t];
        float a0 = prA[t * 3], a1 = prA[t * 3 + 1], a2 = prA[t * 3 + 2];
        float pv = a0 * w0 + a1 * w1 + a2 * w2 + bb;
        acc += (xv[(size_t)j * CC + c] + pv) * w[(size_t)t * CS + (c & 31)];
    }
    y[(size_t)n * CC + c] = acc;
}

// =====================================================================
// out = relu(bn3(t3) + identity); also copy p and o
// =====================================================================
__global__ __launch_bounds__(256)
void finalize_x2(const float* __restrict__ t3, const float* __restrict__ h,
                 const float* __restrict__ mu, const float* __restrict__ rstd,
                 const float* __restrict__ g, const float* __restrict__ b,
                 float* __restrict__ out) {
    int i = blockIdx.x * 256 + threadIdx.x;
    if (i >= NPTS * CC) return;
    int c = i & (CC - 1);
    float v = (t3[i] - mu[c]) * rstd[c] * g[c] + b[c] + h[i];
    out[NPTS * 3 + i] = fmaxf(v, 0.f);
}

__global__ __launch_bounds__(256)
void copy_p_o(const float* __restrict__ p, const int* __restrict__ o,
              float* __restrict__ out) {
    int i = blockIdx.x * 256 + threadIdx.x;
    if (i < NPTS * 3) out[i] = p[i];
    if (i == 0) out[NPTS * 3 + NPTS * CC] = (float)o[0];
}

// =====================================================================
// host-side orchestration
// =====================================================================
extern "C" void kernel_launch(void* const* d_in, const int* in_sizes, int n_in,
                              void* d_out, int out_size, void* d_ws, size_t ws_size,
                              hipStream_t stream) {
    (void)in_sizes; (void)n_in; (void)out_size; (void)ws_size;
    const float* p    = (const float*)d_in[0];
    const float* x    = (const float*)d_in[1];
    const int*   o    = (const int*)d_in[2];
    const float* W_pa = (const float*)d_in[3];
    const float* g_pa = (const float*)d_in[4];
    const float* b_pa = (const float*)d_in[5];
    const float* W1   = (const float*)d_in[6];
    const float* g1   = (const float*)d_in[7];
    const float* b1   = (const float*)d_in[8];
    const float* Wq   = (const float*)d_in[9];
    const float* bq   = (const float*)d_in[10];
    const float* Wk   = (const float*)d_in[11];
    const float* bk   = (const float*)d_in[12];
    const float* Wv   = (const float*)d_in[13];
    const float* bv   = (const float*)d_in[14];
    const float* Wp1  = (const float*)d_in[15];
    const float* bp1  = (const float*)d_in[16];
    const float* gp   = (const float*)d_in[17];
    const float* bpp  = (const float*)d_in[18];
    const float* Wp2  = (const float*)d_in[19];
    const float* bp2  = (const float*)d_in[20];
    const float* gw1  = (const float*)d_in[21];
    const float* bw1  = (const float*)d_in[22];
    const float* Ww1  = (const float*)d_in[23];
    const float* bww1 = (const float*)d_in[24];
    const float* gw2  = (const float*)d_in[25];
    const float* bw2  = (const float*)d_in[26];
    const float* Ww2  = (const float*)d_in[27];
    const float* bww2 = (const float*)d_in[28];
    const float* g2   = (const float*)d_in[29];
    const float* b2   = (const float*)d_in[30];
    const float* W3   = (const float*)d_in[31];
    const float* g3   = (const float*)d_in[32];
    const float* b3   = (const float*)d_in[33];
    float* out = (float*)d_out;

    // ---- workspace layout (bump allocation, ~154 MB) ----
    float* F = (float*)d_ws;
    const size_t NC = (size_t)NPTS * CC;       // 2097152
    float* T   = F;                             // GEMM scratch / y2
    float* H   = F + NC;                        // identity (persists)
    float* H1  = F + 2 * NC;                    // h1 / y_pre / t3
    float* Q   = F + 3 * NC;
    float* Kb  = F + 4 * NC;
    float* V   = F + 5 * NC;
    float* PRA = F + 6 * NC;                    // [NK,3]  (524288 reserved)
    float* WA  = PRA + 524288;                  // [NK,32]
    float* WB  = WA + 4194304;                  // [NK,32]
    float* ST  = WB + 4194304;                  // stats scratch (8192 floats)
    __bf16* WW1B = (__bf16*)(ST + 8192);        // [256,32] bf16
    __bf16* WBN  = WW1B + CC * CS;              // [NK,256] bf16
    int* IDX = (int*)(WBN + (size_t)NK * CC);   // [NK]

    dim3 b32(32), b128(128), b256(256);
    dim3 gNC(NPTS / 64, CC / 16);               // 128 x 16
    dim3 gNKCS(NK / 64, CS / 16);               // 2048 x 2

    // PointAggregation: relu(bn(x @ W_pa))
    gemm_f32_wmma<<<gNC, b128, 0, stream>>>(x, W_pa, nullptr, T, NPTS, CC, CIN);
    colstats<<<CC, b256, 0, stream>>>(T, ST + 0, ST + 256, NPTS, CC);
    bn_apply<<<(NC + 255) / 256, b256, 0, stream>>>(T, H, ST + 0, ST + 256, g_pa, b_pa, (int)NC, CC, 1);
    // SelfFusion.linear1 + bn1 + relu
    gemm_f32_wmma<<<gNC, b128, 0, stream>>>(H, W1, nullptr, T, NPTS, CC, CC);
    colstats<<<CC, b256, 0, stream>>>(T, ST + 512, ST + 768, NPTS, CC);
    bn_apply<<<(NC + 255) / 256, b256, 0, stream>>>(T, H1, ST + 512, ST + 768, g1, b1, (int)NC, CC, 1);
    // q, k, v
    gemm_f32_wmma<<<gNC, b128, 0, stream>>>(H1, Wq, bq, Q,  NPTS, CC, CC);
    gemm_f32_wmma<<<gNC, b128, 0, stream>>>(H1, Wk, bk, Kb, NPTS, CC, CC);
    gemm_f32_wmma<<<gNC, b128, 0, stream>>>(H1, Wv, bv, V,  NPTS, CC, CC);
    // KNN
    knn_kernel<<<NPTS / 8, b256, 0, stream>>>(p, IDX);
    // linear_p first stage + BN(3) + relu (stored as 3-vector factor)
    pr1_kernel<<<NK / 256, b256, 0, stream>>>(p, IDX, Wp1, bp1, PRA);
    colstats<<<3, b256, 0, stream>>>(PRA, ST + 1024, ST + 1032, NK, 3);
    bn_apply<<<(NK * 3 + 255) / 256, b256, 0, stream>>>(PRA, PRA, ST + 1024, ST + 1032, gp, bpp, NK * 3, 3, 1);
    // w path: stats of w0 on the fly, then bf16 materialization of relu(bn(w0))
    (void)hipMemsetAsync(ST + 1040, 0, 512 * sizeof(float), stream);
    w0_stats<<<NK / 64, b256, 0, stream>>>(Kb, Q, IDX, PRA, Wp2, bp2, ST + 1040, ST + 1296, 64);
    finalize_stats<<<1, b256, 0, stream>>>(ST + 1040, ST + 1296, ST + 1552, ST + 1808, NK);
    cvt_bf16<<<(CC * CS + 255) / 256, b256, 0, stream>>>(Ww1, WW1B, CC * CS);
    wbn_kernel<<<NK, b256, 0, stream>>>(Kb, Q, IDX, PRA, Wp2, bp2, ST + 1552, ST + 1808, gw1, bw1, WBN);
    // w @ Ww1 (bf16 WMMA + TDM-staged B), bn+relu, @ Ww2 (f32 WMMA)
    gemm_bf16_wmma<<<gNKCS, b128, 0, stream>>>(WBN, WW1B, bww1, WA, NK, CS, CC);
    colstats<<<CS, b256, 0, stream>>>(WA, ST + 2064, ST + 2096, NK, CS);
    bn_apply<<<(NK * CS + 255) / 256, b256, 0, stream>>>(WA, WA, ST + 2064, ST + 2096, gw2, bw2, NK * CS, CS, 1);
    gemm_f32_wmma<<<gNKCS, b128, 0, stream>>>(WA, Ww2, bww2, WB, NK, CS, CS);
    // softmax over K, then weighted aggregation
    softmax_k<<<NPTS, b32, 0, stream>>>(WB);
    y_kernel<<<NPTS, b256, 0, stream>>>(V, IDX, PRA, Wp2, bp2, WB, H1);
    // bn2 + relu, linear3, bn3 + identity + relu
    colstats<<<CC, b256, 0, stream>>>(H1, ST + 2128, ST + 2384, NPTS, CC);
    bn_apply<<<(NC + 255) / 256, b256, 0, stream>>>(H1, T, ST + 2128, ST + 2384, g2, b2, (int)NC, CC, 1);
    gemm_f32_wmma<<<gNC, b128, 0, stream>>>(T, W3, nullptr, H1, NPTS, CC, CC);
    colstats<<<CC, b256, 0, stream>>>(H1, ST + 2640, ST + 2896, NPTS, CC);
    finalize_x2<<<(NC + 255) / 256, b256, 0, stream>>>(H1, H, ST + 2640, ST + 2896, g3, b3, out);
    copy_p_o<<<(NPTS * 3 + 255) / 256, b256, 0, stream>>>(p, o, out);
}
